// RotationalAttention_6597069767306
// MI455X (gfx1250) — matrix operand order repaired and checked by reading, not compiled, not run on hardware
//
#include <hip/hip_runtime.h>

// ---------------------------------------------------------------------------
// Types for CDNA5 WMMA (wave32): v16bf A/B fragments, v8f accumulators.
// ---------------------------------------------------------------------------
typedef __bf16 bf16;
typedef __bf16 bf16x8  __attribute__((ext_vector_type(8)));
typedef __bf16 bf16x16 __attribute__((ext_vector_type(16)));
typedef float  f32x8   __attribute__((ext_vector_type(8)));
typedef int    i32x4   __attribute__((ext_vector_type(4)));

#if __has_builtin(__builtin_amdgcn_global_load_async_to_lds_b128)
#define HAVE_ASYNC_LDS 1
#endif

__device__ __forceinline__ void wait_async0() {
#if __has_builtin(__builtin_amdgcn_s_wait_asynccnt)
  __builtin_amdgcn_s_wait_asynccnt(0);
#else
  asm volatile("s_wait_asynccnt 0x0" ::: "memory");
#endif
}

#ifdef HAVE_ASYNC_LDS
// GLOBAL_LOAD_ASYNC_TO_LDS_B128: per-lane 16B global -> LDS, tracked by ASYNCcnt.
// Builtin signature (from compiler diagnostics): (v4i AS1*, v4i AS3*, imm, cpol).
__device__ __forceinline__ void async_ld_b128(const void* gsrc, void* ldst) {
  __builtin_amdgcn_global_load_async_to_lds_b128(
      (__attribute__((address_space(1))) i32x4*)gsrc,
      (__attribute__((address_space(3))) i32x4*)ldst, 0, 0);
}
#endif

__device__ __forceinline__ bf16 f2bf(float f) {
  unsigned u = __builtin_bit_cast(unsigned, f);
  u += 0x7fffu + ((u >> 16) & 1u);              // round-to-nearest-even
  unsigned short s = (unsigned short)(u >> 16);
  return __builtin_bit_cast(bf16, s);
}

// A-fragment: 16 contiguous bf16 (K = half*16 + 0..15)
__device__ __forceinline__ bf16x16 ld_split(const bf16* p) {
  bf16x8 lo = *(const bf16x8*)p;
  bf16x8 hi = *(const bf16x8*)(p + 8);
  bf16x16 v;
#pragma unroll
  for (int i = 0; i < 8; ++i) { v[i] = lo[i]; v[i + 8] = hi[i]; }
  return v;
}

// B-fragment: 16 contiguous bf16 (K = half*16 + 0..15)
__device__ __forceinline__ bf16x16 ld_cont(const bf16* p) {
  bf16x8 lo = *(const bf16x8*)p;
  bf16x8 hi = *(const bf16x8*)(p + 8);
  bf16x16 v;
#pragma unroll
  for (int i = 0; i < 8; ++i) { v[i] = lo[i]; v[i + 8] = hi[i]; }
  return v;
}

__device__ __forceinline__ f32x8 wmma_bf16(bf16x16 a, bf16x16 b, f32x8 c) {
  // D = A(16x32 bf16) * B(32x16 bf16) + C(16x16 f32)
  return __builtin_amdgcn_wmma_f32_16x16x32_bf16(false, a, false, b,
                                                 (short)0, c, false, false);
}

// ---------------------------------------------------------------------------
// Kernel 1: fp32 -> bf16 convert
// ---------------------------------------------------------------------------
__global__ __launch_bounds__(256) void k_f32_to_bf16(const float* __restrict__ in,
                                                     bf16* __restrict__ out, int n) {
  int i = blockIdx.x * 256 + threadIdx.x;
  if (i < n) out[i] = f2bf(in[i]);
}

// ---------------------------------------------------------------------------
// Kernel 2: GEMM  C[M][N] = A[M][K] * B[N][K]^T (+bias), bf16 in / fp32 out.
// Block = 256 threads = 8 waves (2 x 4) -> 64 x 256 block tile.
// Wave tile = 32 x 64 = 2x4 WMMA tiles; K step 32 -> 8 WMMA per 12 B128 loads.
// A/B streams prefetched 8 K-steps ahead (global_prefetch_b8).
// ---------------------------------------------------------------------------
__global__ __launch_bounds__(256) void k_gemm_bf16(const bf16* __restrict__ A,
                                                   const bf16* __restrict__ Bm,
                                                   const float* __restrict__ bias,
                                                   float* __restrict__ C,
                                                   int M, int N, int K) {
  const int lane = threadIdx.x & 31;
  const int wave = threadIdx.x >> 5;
  const int half = lane >> 4;
  const int r    = lane & 15;
  const int m0 = blockIdx.y * 64  + (wave >> 2) * 32;
  const int n0 = blockIdx.x * 256 + (wave & 3) * 64;

  f32x8 acc[2][4] = {};
  for (int kk = 0; kk < K; kk += 32) {
    bf16x16 a[2], b[4];
#pragma unroll
    for (int mi = 0; mi < 2; ++mi) {
      const bf16* p = A + (size_t)(m0 + mi * 16 + r) * K + kk + half * 16;
      a[mi] = ld_split(p);
      __builtin_prefetch(p + 256, 0, 3);   // 8 K-steps ahead
    }
#pragma unroll
    for (int ni = 0; ni < 4; ++ni) {
      const bf16* p = Bm + (size_t)(n0 + ni * 16 + r) * K + kk + half * 16;
      b[ni] = ld_cont(p);
      __builtin_prefetch(p + 256, 0, 3);
    }
#pragma unroll
    for (int mi = 0; mi < 2; ++mi)
#pragma unroll
      for (int ni = 0; ni < 4; ++ni)
        acc[mi][ni] = wmma_bf16(a[mi], b[ni], acc[mi][ni]);
  }

#pragma unroll
  for (int mi = 0; mi < 2; ++mi)
#pragma unroll
    for (int ni = 0; ni < 4; ++ni) {
      int col = n0 + ni * 16 + r;
      float bb = bias ? bias[col] : 0.0f;
#pragma unroll
      for (int rr = 0; rr < 8; ++rr) {
        int row = m0 + mi * 16 + rr + half * 8;
        C[(size_t)row * N + col] = acc[mi][ni][rr] + bb;
      }
    }
}

// ---------------------------------------------------------------------------
// Kernel 3: split QKV + RoPE + bf16 head layout.
//  qh,kh: [b][h][t][64] (row-major per head)   vh: [b][h][64][t] (feature-major)
// rotate_half: j<32 -> -x[2j+1]*sin ; j>=32 -> +x[2(j-32)]*sin ; cos/sin idx j&31
// ---------------------------------------------------------------------------
__global__ __launch_bounds__(256) void k_rope_split(const float* __restrict__ qkv,
                                                    bf16* __restrict__ qh,
                                                    bf16* __restrict__ kh,
                                                    bf16* __restrict__ vh) {
  int idx = blockIdx.x * 256 + threadIdx.x;   // B*T*D = 4194304 exactly
  int j = idx & 63;
  int h = (idx >> 6) & 15;
  int t = (idx >> 10) & 2047;
  int b = idx >> 21;

  const float* row = qkv + ((size_t)b * 2048 + t) * 3072;
  int col = h * 64 + j;
  float qv = row[col];
  float kv = row[1024 + col];
  float vv = row[2048 + col];

  int i = j & 31;
  // inv_freq = 10000^(-i/32) = exp(-i * ln(10000)/32)
  float angle = (float)t * __expf(-0.2878231366242558f * (float)i);
  float s, c;
  __sincosf(angle, &s, &c);

  int   p   = (j < 32) ? (2 * j + 1) : (2 * (j - 32));
  float sgn = (j < 32) ? -1.0f : 1.0f;
  float qr = sgn * row[h * 64 + p];
  float kr = sgn * row[1024 + h * 64 + p];

  size_t bh = (size_t)(b * 16 + h);
  qh[(bh * 2048 + t) * 64 + j] = f2bf(qv * c + qr * s);
  kh[(bh * 2048 + t) * 64 + j] = f2bf(kv * c + kr * s);
  vh[(bh * 64 + j) * 2048 + t] = f2bf(vv);
}

// ---------------------------------------------------------------------------
// Kernel 4: flash attention, hd=64, T=2048, scale = 1/8.
// Block = 128 threads = 4 waves; block handles one (b,h) and 64 q rows
// (16 q rows per wave). Key-block of 32 per iteration:
//   S(16x32) = Q K^T : 4 WMMA; online softmax; O += P V : 4 WMMA.
// K / V^T tiles staged into LDS with GLOBAL_LOAD_ASYNC_TO_LDS_B128 when the
// toolchain exposes the builtin (ASYNCcnt path), else manual staging.
// ---------------------------------------------------------------------------
__global__ __launch_bounds__(128) void k_attention(const bf16* __restrict__ qh,
                                                   const bf16* __restrict__ kh,
                                                   const bf16* __restrict__ vh,
                                                   bf16* __restrict__ out) {
  __shared__ __attribute__((aligned(16))) bf16 lK[32 * 64];    // [n][d]
  __shared__ __attribute__((aligned(16))) bf16 lV[64 * 32];    // [d][n]  (V^T tile)
  __shared__ __attribute__((aligned(16))) bf16 lP[4][16 * 32]; // per-wave P

  const int lane = threadIdx.x & 31;
  const int wave = threadIdx.x >> 5;
  const int half = lane >> 4;
  const int r    = lane & 15;

  const int bh  = blockIdx.x >> 5;  // 0..31
  const int qsb = blockIdx.x & 31;  // 0..31
  const int qb  = qsb * 64 + wave * 16;
  const int b   = bh >> 4;
  const int h   = bh & 15;

  const bf16* Qb = qh + (size_t)bh * 2048 * 64;
  const bf16* Kb = kh + (size_t)bh * 2048 * 64;
  const bf16* Vb = vh + (size_t)bh * 64 * 2048;

  // per-thread staging coordinates (16 bf16 each for lK and lV)
  const int e   = threadIdx.x * 16;
  const int krr = e >> 6, kdd = e & 63;
  const int vrr = e >> 5, vcc = e & 31;

  // Resident Q fragments (two K-halves of hd=64)
  bf16x16 aQ[2];
#pragma unroll
  for (int dh = 0; dh < 2; ++dh)
    aQ[dh] = ld_split(Qb + (size_t)(qb + r) * 64 + dh * 32 + half * 16);

  f32x8 O[4] = {};
  float mrow[8], lrow[8];
#pragma unroll
  for (int i = 0; i < 8; ++i) { mrow[i] = -1e30f; lrow[i] = 0.0f; }

  for (int t0 = 0; t0 < 2048; t0 += 32) {
    __syncthreads();
    {
      const bf16* ksrc = Kb + (size_t)(t0 + krr) * 64 + kdd;
      const bf16* vsrc = Vb + (size_t)vrr * 2048 + t0 + vcc;
#ifdef HAVE_ASYNC_LDS
      async_ld_b128(ksrc,     &lK[krr * 64 + kdd]);
      async_ld_b128(ksrc + 8, &lK[krr * 64 + kdd + 8]);
      async_ld_b128(vsrc,     &lV[vrr * 32 + vcc]);
      async_ld_b128(vsrc + 8, &lV[vrr * 32 + vcc + 8]);
      wait_async0();
#else
      *(bf16x8*)&lK[krr * 64 + kdd]     = *(const bf16x8*)ksrc;
      *(bf16x8*)&lK[krr * 64 + kdd + 8] = *(const bf16x8*)(ksrc + 8);
      *(bf16x8*)&lV[vrr * 32 + vcc]     = *(const bf16x8*)vsrc;
      *(bf16x8*)&lV[vrr * 32 + vcc + 8] = *(const bf16x8*)(vsrc + 8);
#endif
    }
    __syncthreads();

    // S = Q K^T  (two 16-key tiles, K-dim 64 = 2 WMMA each)
    f32x8 S0 = {}, S1 = {};
#pragma unroll
    for (int dh = 0; dh < 2; ++dh) {
      S0 = wmma_bf16(aQ[dh], ld_cont(&lK[(r)      * 64 + dh * 32 + half * 16]), S0);
      S1 = wmma_bf16(aQ[dh], ld_cont(&lK[(16 + r) * 64 + dh * 32 + half * 16]), S1);
    }

    // Online softmax over this 32-key block. C-layout: lane half owns 8 rows.
    float alpha[8];
#pragma unroll
    for (int rr = 0; rr < 8; ++rr) {
      float s0 = S0[rr] * 0.125f, s1 = S1[rr] * 0.125f;
      float mx = fmaxf(s0, s1);
      mx = fmaxf(mx, __shfl_xor(mx, 1));
      mx = fmaxf(mx, __shfl_xor(mx, 2));
      mx = fmaxf(mx, __shfl_xor(mx, 4));
      mx = fmaxf(mx, __shfl_xor(mx, 8));
      float mnew = fmaxf(mrow[rr], mx);
      float a  = __expf(mrow[rr] - mnew);
      float p0 = __expf(s0 - mnew);
      float p1 = __expf(s1 - mnew);
      float ps = p0 + p1;
      ps += __shfl_xor(ps, 1);
      ps += __shfl_xor(ps, 2);
      ps += __shfl_xor(ps, 4);
      ps += __shfl_xor(ps, 8);
      lrow[rr] = lrow[rr] * a + ps;
      mrow[rr] = mnew;
      alpha[rr] = a;
      int m = rr + half * 8;                      // C-layout row this lane owns
      lP[wave][m * 32 + r]      = f2bf(p0);
      lP[wave][m * 32 + 16 + r] = f2bf(p1);
    }
#pragma unroll
    for (int dt = 0; dt < 4; ++dt)
#pragma unroll
      for (int rr = 0; rr < 8; ++rr) O[dt][rr] *= alpha[rr];

    // P back as A-fragment (K=32). Per-wave LDS; DS ops are in-order per wave.
    bf16x16 aP = ld_split(&lP[wave][r * 32 + half * 16]);
#pragma unroll
    for (int dt = 0; dt < 4; ++dt) {
      bf16x16 bv = ld_cont(&lV[(dt * 16 + r) * 32 + half * 16]);
      O[dt] = wmma_bf16(aP, bv, O[dt]);
    }
  }

  // Normalize and write bf16 into [b][t][1024] layout for the output GEMM.
#pragma unroll
  for (int rr = 0; rr < 8; ++rr) {
    float inv = 1.0f / lrow[rr];
    int t = qb + rr + half * 8;
#pragma unroll
    for (int dt = 0; dt < 4; ++dt)
      out[((size_t)(b * 2048 + t)) * 1024 + h * 64 + dt * 16 + r] =
          f2bf(O[dt][rr] * inv);
  }
}

// ---------------------------------------------------------------------------
// Host-side pipeline. Workspace layout (bytes), ~88 MB total:
//   [0,        8M)  x_bf16        (dead after gemm1)
//   [8M,      14M)  w_qkv_bf16
//   [14M,     16M)  w_out_bf16
//   [16M,     64M)  qkv_f32       (dead after rope; attn_bf16 aliases here)
//   [64M,     72M)  qh bf16
//   [72M,     80M)  kh bf16
//   [80M,     88M)  vh bf16 (transposed)
// ---------------------------------------------------------------------------
extern "C" void kernel_launch(void* const* d_in, const int* in_sizes, int n_in,
                              void* d_out, int out_size, void* d_ws, size_t ws_size,
                              hipStream_t stream) {
  (void)in_sizes; (void)n_in; (void)out_size; (void)ws_size;
  const float* x     = (const float*)d_in[0];
  const float* w_qkv = (const float*)d_in[1];
  const float* w_out = (const float*)d_in[2];
  const float* b_out = (const float*)d_in[3];
  float* out = (float*)d_out;
  char* ws = (char*)d_ws;

  bf16*  xb    = (bf16*)(ws);
  bf16*  wqkvb = (bf16*)(ws + 8388608);
  bf16*  woutb = (bf16*)(ws + 14680064);
  float* qkv   = (float*)(ws + 16777216);
  bf16*  attnb = (bf16*)(ws + 16777216);   // alias: qkv_f32 dead after rope
  bf16*  qh    = (bf16*)(ws + 67108864);
  bf16*  kh    = (bf16*)(ws + 75497472);
  bf16*  vh    = (bf16*)(ws + 83886080);

  k_f32_to_bf16<<<(4194304 + 255) / 256, 256, 0, stream>>>(x, xb, 4194304);
  k_f32_to_bf16<<<(3145728 + 255) / 256, 256, 0, stream>>>(w_qkv, wqkvb, 3145728);
  k_f32_to_bf16<<<(1048576 + 255) / 256, 256, 0, stream>>>(w_out, woutb, 1048576);

  // qkv[4096][3072] = x[4096][1024] @ w_qkv[3072][1024]^T
  k_gemm_bf16<<<dim3(12, 64), 256, 0, stream>>>(xb, wqkvb, nullptr, qkv,
                                                4096, 3072, 1024);
  // RoPE + head split
  k_rope_split<<<16384, 256, 0, stream>>>(qkv, qh, kh, vh);
  // attention: 32 (b,h) x 32 q-superblocks
  k_attention<<<1024, 128, 0, stream>>>(qh, kh, vh, attnb);
  // out[4096][1024] = attn[4096][1024] @ w_out[1024][1024]^T + b_out
  k_gemm_bf16<<<dim3(4, 64), 256, 0, stream>>>(attnb, woutb, b_out, out,
                                               4096, 1024, 1024);
}
